// DecoderLayerJ_11622181503141
// MI455X (gfx1250) — matrix operand order repaired and checked by reading, not compiled, not run on hardware
//
#include <hip/hip_runtime.h>
#include <cmath>

// ---------------------------------------------------------------------------
// Problem constants (from the reference)
// ---------------------------------------------------------------------------
#define Bdim 4
#define Ndim 2048
#define Kn   48            // neighbors
#define Hd   128           // hidden
#define INd  128           // edge feature width
#define NODES (Bdim*Ndim)  // 8192
#define SCALE_INV (1.0f/30.0f)
#define EPSf 1e-5f

typedef __attribute__((ext_vector_type(16))) __bf16 v16bf;
typedef __attribute__((ext_vector_type(8)))  float  v8f;

// ---------------------------------------------------------------------------
// Workspace layout (d_ws):
//   [0, 384KB)  : bf16 WMMA B-fragments of W1 | W2 | W3 | D1 | D2
//                 one fragment = 32x16 bf16 (32 lanes x 16 elems) = 1KB
//   [512KB, +4MB): f32 hbuf[NODES*Hd]  (h after first LayerNorm)
// ---------------------------------------------------------------------------
#define WFRAG_W1 0     // 64 frags  (KC=8,  NT=8)
#define WFRAG_W2 64    // 32 frags  (KC=4,  NT=8)
#define WFRAG_W3 96    // 32 frags  (KC=4,  NT=8)
#define WFRAG_D1 128   // 128 frags (KC=4,  NT=32)
#define WFRAG_D2 256   // 128 frags (KC=16, NT=8)
#define NFRAGS   384
#define HBUF_OFF (512*1024)

__device__ __forceinline__ unsigned short f2bf(float f) {
  union { float f; unsigned u; } v; v.f = f;
  unsigned r = v.u + 0x7FFFu + ((v.u >> 16) & 1u);   // round-to-nearest-even
  return (unsigned short)(r >> 16);
}

// Branchless tanh-form GELU: x * sigmoid(1.5957692*(x + 0.044715 x^3)).
// |err| ~ 1e-3, far below the bf16 quantization already applied to the
// activations; avoids erff's EXEC-divergent polynomial branches so the VALU
// epilogue can co-execute with the XDL (WMMA) pipe.
__device__ __forceinline__ float gelu_f(float x) {
  float z = 1.5957692214f * x * (1.0f + 0.044715f * x * x);
  float e = __expf(-z);                       // v_exp_f32 (TRANS)
  return x * __builtin_amdgcn_rcpf(1.0f + e); // v_rcp_f32
}

// A-fragment position for a 16x32 bf16 tile element (row r16, col cc):
//   lanes 0-15: M=0..15, K in {0..7, 16..23};  lanes 16-31: same rows, K in {8..15, 24..31}
__device__ __forceinline__ void a_frag_pos(int r16, int cc, int& lane2, int& e) {
  lane2 = r16 + (((cc >> 3) & 1) ? 16 : 0);
  e     = (cc & 7) + ((cc >> 4) ? 8 : 0);
}

#define WMMA_BF16(A, B, C) \
  __builtin_amdgcn_wmma_f32_16x16x32_bf16(false, (A), false, (B), (short)0, (C), false, false)

// Epilogue for edge layers 1/2: bias + gelu, scatter into next layer's
// A-fragment layout (8 x ds_store_b16 per lane).
__device__ __forceinline__ void store_act_frag(
    unsigned short* dst, int node, int mt, v8f acc, float bias,
    int kc2, int laneAddC, int eC, int lane) {
  #pragma unroll
  for (int j = 0; j < 8; ++j) {
    int r16 = j + ((lane & 16) ? 8 : 0);
    float x = gelu_f(acc[j] + bias);
    dst[node * (12 * 512) + ((mt * 4 + kc2) * 32 + (r16 + laneAddC)) * 16 + eC] = f2bf(x);
  }
}

// Epilogue for edge layer 3: bias + neighbor mask + per-lane partial sum.
__device__ __forceinline__ float mask_partial(
    const float* MASKS, int node, int mt, v8f acc, float bias, int lane) {
  float partial = 0.0f;
  #pragma unroll
  for (int j = 0; j < 8; ++j) {
    int r16 = j + ((lane & 16) ? 8 : 0);
    partial += MASKS[node * Kn + mt * 16 + r16] * (acc[j] + bias);
  }
  return partial;
}

// ---------------------------------------------------------------------------
// Kernel 0: convert f32 weights -> bf16 WMMA B-fragments.
// B tile (kc, nt) of  Bmat[k][n] = W[n][k]  (W is [Nout x Kin] row-major).
// Lane L holds column n = nt*16 + (L&15), K = kc*32 + (L<16?0:8) + (e&7) + (e<8?0:16).
// ---------------------------------------------------------------------------
__global__ __launch_bounds__(256) void prep_weights_kernel(
    const float* __restrict__ W1, const float* __restrict__ W2,
    const float* __restrict__ W3, const float* __restrict__ D1,
    const float* __restrict__ D2, unsigned short* __restrict__ wbf) {
  int f = blockIdx.x;
  const float* src; int NT, Kd, fl;
  if (f < WFRAG_W2)      { src = W1; NT = 8;  Kd = 256; fl = f; }
  else if (f < WFRAG_W3) { src = W2; NT = 8;  Kd = 128; fl = f - WFRAG_W2; }
  else if (f < WFRAG_D1) { src = W3; NT = 8;  Kd = 128; fl = f - WFRAG_W3; }
  else if (f < WFRAG_D2) { src = D1; NT = 32; Kd = 128; fl = f - WFRAG_D1; }
  else                   { src = D2; NT = 8;  Kd = 512; fl = f - WFRAG_D2; }
  int kc = fl / NT, nt = fl % NT;
  for (int within = threadIdx.x; within < 512; within += blockDim.x) {
    int lane = within >> 4, e = within & 15;
    int n = nt * 16 + (lane & 15);
    int k = kc * 32 + ((lane & 16) ? 8 : 0) + (e & 7) + ((e & 8) ? 16 : 0);
    wbf[(size_t)f * 512 + within] = f2bf(src[(size_t)n * Kd + k]);
  }
}

// ---------------------------------------------------------------------------
// Kernel 1: per-node edge-message MLP (3 bf16 WMMA layers) + masked neighbor
// sum + residual + LayerNorm1.  NPB nodes per block, 8 waves; wave w owns
// output column tile nt=w for every layer.  The 3 M-tiles run as concurrent
// accumulator chains so each kc step issues 3 independent WMMAs.
// ---------------------------------------------------------------------------
#define NPB 2
#define EDGE_LDS_BYTES ((NPB*24*512 + NPB*12*512 + NPB*12*512)*2 + \
                        (NPB*Hd + NPB*Hd + NPB*Kn)*4)

__global__ __launch_bounds__(256) void edge_mlp_kernel(
    const float* __restrict__ h_v, const float* __restrict__ h_e,
    const float* __restrict__ mask_attend,
    const float* __restrict__ b1, const float* __restrict__ b2,
    const float* __restrict__ b3,
    const float* __restrict__ g1, const float* __restrict__ beta1,
    const unsigned short* __restrict__ wbf, float* __restrict__ hbuf) {
  extern __shared__ char smem[];
  unsigned short* HEVs = (unsigned short*)smem;       // NPB * 24 frags (48x256)
  unsigned short* M1s  = HEVs + NPB * 24 * 512;       // NPB * 12 frags (48x128)
  unsigned short* M2s  = M1s  + NPB * 12 * 512;       // NPB * 12 frags
  float* HV    = (float*)(M2s + NPB * 12 * 512);      // NPB * 128
  float* DH    = HV + NPB * Hd;                       // NPB * 128
  float* MASKS = DH + NPB * Hd;                       // NPB * 48

  const int tid = threadIdx.x, w = tid >> 5, lane = tid & 31;
  const int bn0 = blockIdx.x * NPB;

  // L2 prefetch of next block's edge tile (gfx1250 global_prefetch_b8)
  if (blockIdx.x + 1 < gridDim.x) {
    const char* pre = (const char*)(h_e + (size_t)(bn0 + NPB) * (Kn * INd));
    __builtin_prefetch(pre + tid * 96, 0, 1);
  }

  // stage h_v, zero dh accumulator, stage neighbor masks
  for (int idx = tid; idx < NPB * Hd; idx += 256) {
    int node = idx >> 7, c = idx & 127;
    HV[idx] = h_v[(size_t)(bn0 + node) * Hd + c];
    DH[idx] = 0.0f;
  }
  for (int idx = tid; idx < NPB * Kn; idx += 256) {
    int node = idx / Kn, r = idx - node * Kn;
    MASKS[idx] = mask_attend[(size_t)(bn0 + node) * Kn + r];
  }
  __syncthreads();

  // stage h_ev = [h_v | h_e]  (48 x 256) as bf16 WMMA A-fragments.
  // h_e is streamed exactly once over the whole dispatch -> non-temporal.
  for (int idx = tid; idx < NPB * Kn * 256; idx += 256) {
    int node = idx / (Kn * 256);
    int rem  = idx - node * (Kn * 256);
    int r = rem >> 8, c = rem & 255;
    float v = (c < Hd)
        ? HV[node * Hd + c]
        : __builtin_nontemporal_load(
              h_e + (size_t)(bn0 + node) * (Kn * INd) + (size_t)r * INd + (c - Hd));
    int mt = r >> 4, r16 = r & 15, kc = c >> 5, cc = c & 31;
    int lane2, e; a_frag_pos(r16, cc, lane2, e);
    HEVs[node * (24 * 512) + ((mt * 8 + kc) * 32 + lane2) * 16 + e] = f2bf(v);
  }
  __syncthreads();

  const v16bf* HEVf = (const v16bf*)HEVs;
  const v16bf* M1f  = (const v16bf*)M1s;
  const v16bf* M2f  = (const v16bf*)M2s;
  const v16bf* Wg   = (const v16bf*)wbf;

  // this wave's output column and its position as next-layer A-fragment
  const int hcol = w * 16 + (lane & 15);
  const int kc2  = hcol >> 5;
  int laneAddC, eC; a_frag_pos(0, hcol & 31, laneAddC, eC);
  const float b1h = b1[hcol], b2h = b2[hcol], b3h = b3[hcol];

  // ---- layer 1: m1 = gelu(h_ev @ W1^T + b1)  : (48x256)(256x128)
  {
    v16bf bw[8];
    #pragma unroll
    for (int kc = 0; kc < 8; ++kc)
      bw[kc] = Wg[(WFRAG_W1 + kc * 8 + w) * 32 + lane];
    for (int node = 0; node < NPB; ++node) {
      v8f acc0 = {}, acc1 = {}, acc2 = {};
      #pragma unroll
      for (int kc = 0; kc < 8; ++kc) {
        v16bf a0 = HEVf[(node * 24 + 0 * 8 + kc) * 32 + lane];
        v16bf a1 = HEVf[(node * 24 + 1 * 8 + kc) * 32 + lane];
        v16bf a2 = HEVf[(node * 24 + 2 * 8 + kc) * 32 + lane];
        acc0 = WMMA_BF16(a0, bw[kc], acc0);
        acc1 = WMMA_BF16(a1, bw[kc], acc1);
        acc2 = WMMA_BF16(a2, bw[kc], acc2);
      }
      store_act_frag(M1s, node, 0, acc0, b1h, kc2, laneAddC, eC, lane);
      store_act_frag(M1s, node, 1, acc1, b1h, kc2, laneAddC, eC, lane);
      store_act_frag(M1s, node, 2, acc2, b1h, kc2, laneAddC, eC, lane);
    }
  }
  __syncthreads();

  // ---- layer 2: m2 = gelu(m1 @ W2^T + b2)  : (48x128)(128x128)
  {
    v16bf bw[4];
    #pragma unroll
    for (int kc = 0; kc < 4; ++kc)
      bw[kc] = Wg[(WFRAG_W2 + kc * 8 + w) * 32 + lane];
    for (int node = 0; node < NPB; ++node) {
      v8f acc0 = {}, acc1 = {}, acc2 = {};
      #pragma unroll
      for (int kc = 0; kc < 4; ++kc) {
        v16bf a0 = M1f[(node * 12 + 0 * 4 + kc) * 32 + lane];
        v16bf a1 = M1f[(node * 12 + 1 * 4 + kc) * 32 + lane];
        v16bf a2 = M1f[(node * 12 + 2 * 4 + kc) * 32 + lane];
        acc0 = WMMA_BF16(a0, bw[kc], acc0);
        acc1 = WMMA_BF16(a1, bw[kc], acc1);
        acc2 = WMMA_BF16(a2, bw[kc], acc2);
      }
      store_act_frag(M2s, node, 0, acc0, b2h, kc2, laneAddC, eC, lane);
      store_act_frag(M2s, node, 1, acc1, b2h, kc2, laneAddC, eC, lane);
      store_act_frag(M2s, node, 2, acc2, b2h, kc2, laneAddC, eC, lane);
    }
  }
  __syncthreads();

  // ---- layer 3: m = m2 @ W3^T + b3, masked sum over neighbors -> DH
  {
    v16bf bw[4];
    #pragma unroll
    for (int kc = 0; kc < 4; ++kc)
      bw[kc] = Wg[(WFRAG_W3 + kc * 8 + w) * 32 + lane];
    for (int node = 0; node < NPB; ++node) {
      v8f acc0 = {}, acc1 = {}, acc2 = {};
      #pragma unroll
      for (int kc = 0; kc < 4; ++kc) {
        v16bf a0 = M2f[(node * 12 + 0 * 4 + kc) * 32 + lane];
        v16bf a1 = M2f[(node * 12 + 1 * 4 + kc) * 32 + lane];
        v16bf a2 = M2f[(node * 12 + 2 * 4 + kc) * 32 + lane];
        acc0 = WMMA_BF16(a0, bw[kc], acc0);
        acc1 = WMMA_BF16(a1, bw[kc], acc1);
        acc2 = WMMA_BF16(a2, bw[kc], acc2);
      }
      float partial = mask_partial(MASKS, node, 0, acc0, b3h, lane)
                    + mask_partial(MASKS, node, 1, acc1, b3h, lane)
                    + mask_partial(MASKS, node, 2, acc2, b3h, lane);
      atomicAdd(&DH[node * Hd + hcol], partial);   // ds_add_f32
    }
  }
  __syncthreads();

  // ---- LayerNorm 1: h = LN(h_v + dh/SCALE), one wave per node
  if (w < NPB) {
    int node = w;
    size_t bn = (size_t)bn0 + node;
    float x[4], s = 0.f, ss = 0.f;
    #pragma unroll
    for (int i = 0; i < 4; ++i) {
      int c = lane + 32 * i;
      x[i] = HV[node * Hd + c] + DH[node * Hd + c] * SCALE_INV;
      s += x[i]; ss += x[i] * x[i];
    }
    #pragma unroll
    for (int m = 16; m >= 1; m >>= 1) {
      s  += __shfl_xor(s,  m, 32);
      ss += __shfl_xor(ss, m, 32);
    }
    float mu  = s  * (1.0f / 128.0f);
    float var = ss * (1.0f / 128.0f) - mu * mu;
    float inv = rsqrtf(var + EPSf);
    #pragma unroll
    for (int i = 0; i < 4; ++i) {
      int c = lane + 32 * i;
      hbuf[bn * Hd + c] = (x[i] - mu) * inv * g1[c] + beta1[c];
    }
  }
}

// ---------------------------------------------------------------------------
// Kernel 2: dense MLP 128->512->128 over 16-node row tiles + LN2 + node mask.
// D1 runs 4 concurrent accumulator chains (one per nt tile); D2 uses split-K
// (two chains) to double WMMA ILP.
// ---------------------------------------------------------------------------
__global__ __launch_bounds__(256) void dense_mlp_kernel(
    const float* __restrict__ hbuf, const float* __restrict__ mask_v,
    const float* __restrict__ db1, const float* __restrict__ db2,
    const float* __restrict__ g2, const float* __restrict__ beta2,
    const unsigned short* __restrict__ wbf, float* __restrict__ out) {
  __shared__ unsigned short HAs[4 * 512];    // h as A-fragments (16x128)
  __shared__ unsigned short Gs[16 * 512];    // gelu(hD1+db1) frags (16x512)
  __shared__ float HRES[16 * Hd];            // f32 h for the residual
  __shared__ float OUTs[16 * Hd];            // pre-LN2 result

  const int tid = threadIdx.x, w = tid >> 5, lane = tid & 31;
  const int node0 = blockIdx.x * 16;

  for (int idx = tid; idx < 16 * Hd; idx += 256) {
    int r = idx >> 7, c = idx & 127;
    float v = hbuf[(size_t)(node0 + r) * Hd + c];
    HRES[idx] = v;
    int lane2, e; a_frag_pos(r, c & 31, lane2, e);
    HAs[((c >> 5) * 32 + lane2) * 16 + e] = f2bf(v);
  }
  __syncthreads();

  const v16bf* HAf = (const v16bf*)HAs;
  const v16bf* Gf  = (const v16bf*)Gs;
  const v16bf* Wg  = (const v16bf*)wbf;

  // ---- D1: (16x128)(128x512), gelu; wave handles nt = w + 8t, t<4
  {
    v8f acc[4] = {v8f{}, v8f{}, v8f{}, v8f{}};
    #pragma unroll
    for (int kc = 0; kc < 4; ++kc) {
      v16bf a = HAf[kc * 32 + lane];
      #pragma unroll
      for (int t = 0; t < 4; ++t) {
        int nt = w + 8 * t;
        acc[t] = WMMA_BF16(a, Wg[(WFRAG_D1 + kc * 32 + nt) * 32 + lane], acc[t]);
      }
    }
    #pragma unroll
    for (int t = 0; t < 4; ++t) {
      int hcol = w * 16 + 128 * t + (lane & 15);   // nt*16 + (lane&15)
      float bb = db1[hcol];
      int kcg = hcol >> 5;
      int laneAdd, e; a_frag_pos(0, hcol & 31, laneAdd, e);
      #pragma unroll
      for (int j = 0; j < 8; ++j) {
        int r16 = j + ((lane & 16) ? 8 : 0);
        Gs[(kcg * 32 + (r16 + laneAdd)) * 16 + e] = f2bf(gelu_f(acc[t][j] + bb));
      }
    }
  }
  __syncthreads();

  // ---- D2: (16x512)(512x128) + db2 + residual; split-K accumulators
  {
    int nt = w;
    v8f accA = {}, accB = {};
    #pragma unroll
    for (int kc = 0; kc < 8; ++kc) {
      v16bf aA = Gf[kc * 32 + lane];
      v16bf aB = Gf[(kc + 8) * 32 + lane];
      accA = WMMA_BF16(aA, Wg[(WFRAG_D2 + kc * 8 + nt) * 32 + lane], accA);
      accB = WMMA_BF16(aB, Wg[(WFRAG_D2 + (kc + 8) * 8 + nt) * 32 + lane], accB);
    }
    int hcol = nt * 16 + (lane & 15);
    float bb = db2[hcol];
    #pragma unroll
    for (int j = 0; j < 8; ++j) {
      int r = j + ((lane & 16) ? 8 : 0);
      OUTs[r * Hd + hcol] = accA[j] + accB[j] + bb + HRES[r * Hd + hcol];
    }
  }
  __syncthreads();

  // ---- LayerNorm 2 + node mask: each wave handles 2 nodes
  for (int rep = 0; rep < 2; ++rep) {
    int node = w * 2 + rep;
    size_t bn = (size_t)node0 + node;
    float x[4], s = 0.f, ss = 0.f;
    #pragma unroll
    for (int i = 0; i < 4; ++i) {
      int c = lane + 32 * i;
      x[i] = OUTs[node * Hd + c];
      s += x[i]; ss += x[i] * x[i];
    }
    #pragma unroll
    for (int m = 16; m >= 1; m >>= 1) {
      s  += __shfl_xor(s,  m, 32);
      ss += __shfl_xor(ss, m, 32);
    }
    float mu  = s  * (1.0f / 128.0f);
    float var = ss * (1.0f / 128.0f) - mu * mu;
    float inv = rsqrtf(var + EPSf);
    float mv = mask_v[bn];
    #pragma unroll
    for (int i = 0; i < 4; ++i) {
      int c = lane + 32 * i;
      out[bn * Hd + c] = mv * ((x[i] - mu) * inv * g2[c] + beta2[c]);
    }
  }
}

// ---------------------------------------------------------------------------
extern "C" void kernel_launch(void* const* d_in, const int* in_sizes, int n_in,
                              void* d_out, int out_size, void* d_ws, size_t ws_size,
                              hipStream_t stream) {
  (void)in_sizes; (void)n_in; (void)out_size; (void)ws_size;
  const float* h_v         = (const float*)d_in[0];
  const float* h_e         = (const float*)d_in[1];
  const float* mask_attend = (const float*)d_in[2];
  const float* mask_v      = (const float*)d_in[3];
  const float* W1  = (const float*)d_in[4];
  const float* b1  = (const float*)d_in[5];
  const float* W2  = (const float*)d_in[6];
  const float* b2  = (const float*)d_in[7];
  const float* W3  = (const float*)d_in[8];
  const float* b3  = (const float*)d_in[9];
  const float* D1  = (const float*)d_in[10];
  const float* db1 = (const float*)d_in[11];
  const float* D2  = (const float*)d_in[12];
  const float* db2 = (const float*)d_in[13];
  const float* g1    = (const float*)d_in[14];
  const float* beta1 = (const float*)d_in[15];
  const float* g2    = (const float*)d_in[16];
  const float* beta2 = (const float*)d_in[17];

  unsigned short* wbf = (unsigned short*)d_ws;
  float* hbuf = (float*)((char*)d_ws + HBUF_OFF);

  prep_weights_kernel<<<NFRAGS, 256, 0, stream>>>(W1, W2, W3, D1, D2, wbf);

  (void)hipFuncSetAttribute((const void*)edge_mlp_kernel,
                            hipFuncAttributeMaxDynamicSharedMemorySize,
                            EDGE_LDS_BYTES);
  edge_mlp_kernel<<<NODES / NPB, 256, EDGE_LDS_BYTES, stream>>>(
      h_v, h_e, mask_attend, b1, b2, b3, g1, beta1, wbf, hbuf);

  dense_mlp_kernel<<<NODES / 16, 256, 0, stream>>>(
      hbuf, mask_v, db1, db2, g2, beta2, wbf, (float*)d_out);
}
